// SlowMAF_89464168776457
// MI455X (gfx1250) — matrix operand order repaired and checked by reading, not compiled, not run on hardware
//
#include <hip/hip_runtime.h>
#include <hip/hip_bf16.h>

// ---------------------------------------------------------------------------
// Fused masked-autoregressive-flow forward for MI455X (gfx1250, wave32, WMMA).
//
// Compute-bound (~75 GFLOP vs ~13 MB HBM traffic), so everything is fused:
// intermediates live in LDS as bf16, matmuls run on v_wmma_f32_16x16x32_bf16
// with f32 accumulation. One block = 128 batch rows, loops over all 31
// per-dim MLPs so log_det accumulates in registers (no atomics) and the 5 MB
// of weights stay L2-resident. ~83 KB LDS -> 3 workgroups per 320 KB WGP so
// the weight-load and WMMA phases of co-resident blocks overlap.
// ---------------------------------------------------------------------------

typedef __bf16 bf16_t;
typedef __bf16  bf16x8  __attribute__((ext_vector_type(8)));
typedef __bf16  bf16x16 __attribute__((ext_vector_type(16)));
typedef float   floatx8 __attribute__((ext_vector_type(8)));

#define B_TOT   32768
#define DDIM    32
#define NH      128
#define BM      128      // batch rows per block
#define NTHR    256      // 8 wave32s; each wave owns 16 rows
#define SX      40       // LDS stride for x tile (32 + 8 pad, keeps 16B align)
#define SH      136      // LDS stride for h / W_T (128 + 8 pad)

__device__ __forceinline__ bf16x16 lds_frag(const bf16_t* p0, const bf16_t* p1) {
    bf16x8 lo = *(const bf16x8*)p0;   // ds_load_b128
    bf16x8 hi = *(const bf16x8*)p1;   // ds_load_b128
    return __builtin_shufflevector(lo, hi, 0,1,2,3,4,5,6,7,8,9,10,11,12,13,14,15);
}

__global__ __launch_bounds__(NTHR)
void maf_fused_kernel(const float* __restrict__ x,  const float* __restrict__ p0,
                      const float* __restrict__ W1, const float* __restrict__ b1,
                      const float* __restrict__ W2, const float* __restrict__ b2,
                      const float* __restrict__ W3, const float* __restrict__ b3,
                      const float* __restrict__ W4, const float* __restrict__ b4,
                      float* __restrict__ z, float* __restrict__ logdet)
{
    __shared__ bf16_t xLds[BM * SX];     // x tile, bf16, row-major
    __shared__ bf16_t hLds[BM * SH];     // activations, row-major
    __shared__ bf16_t wtLds[NH * SH];    // W transposed: [n][k]
    __shared__ float  biasLds[NH];       // current layer bias (f32)
    __shared__ float  w4Lds[NH * 2];     // W4[i] interleaved [k][col]
    __shared__ float  w4bLds[2];         // b4[i]
    __shared__ float  stLds[BM * 2];     // (s, t) per row

    const int t        = threadIdx.x;
    const int row0     = blockIdx.x * BM;
    const int lane     = t & 31;
    const int wave     = t >> 5;
    const int l        = lane & 15;      // column/row-within-tile id
    const int hh       = lane >> 4;      // lane-half select
    const int waveRow0 = wave * 16;      // 16 M-rows per wave

    // ---- stage x tile into LDS as bf16 (coalesced) ----
    for (int e = t; e < BM * DDIM; e += NTHR) {
        int r = e >> 5, c = e & 31;
        xLds[r * SX + c] = (bf16_t)x[(size_t)(row0 + r) * DDIM + c];
    }

    float ld_acc = p0[0];                // s for dim 0 (only rows t<128 use it)

    // A-fragment base for this lane (ISA 16-bit A 16x32 layout:
    //   lane<16: row l, K in {0..7, 16..23}; lane>=16: row l, K+8)
    const bf16_t* aX = &xLds[(waveRow0 + l) * SX + hh * 8];
    const bf16_t* aH = &hLds[(waveRow0 + l) * SH + hh * 8];

    floatx8 acc[8];
    const floatx8 vzero = {0.f,0.f,0.f,0.f,0.f,0.f,0.f,0.f};

    // load a K x 128 weight block from global (row-major [k][n]) into
    // wtLds as [n][k] bf16 (float4 reads keep global access coalesced),
    // and stage the layer bias into LDS alongside it.
    auto load_wt = [&](const float* w, int klen, const float* bias) {
        for (int e4 = t * 4; e4 < klen * NH; e4 += NTHR * 4) {
            float4 v = *(const float4*)(w + e4);
            int k = e4 >> 7, n = e4 & 127;
            wtLds[(n + 0) * SH + k] = (bf16_t)v.x;
            wtLds[(n + 1) * SH + k] = (bf16_t)v.y;
            wtLds[(n + 2) * SH + k] = (bf16_t)v.z;
            wtLds[(n + 3) * SH + k] = (bf16_t)v.w;
        }
        if (t < NH) biasLds[t] = bias[t];
    };

    // one WMMA layer: for each K-step, batch-load all 8 B fragments, then
    // issue the 8 WMMAs back-to-back (single s_wait_dscnt, full XDL pipe).
    auto mma = [&](const bf16_t* aBase, int klen) {
        for (int k = 0; k < klen; k += 32) {
            bf16x16 a = lds_frag(aBase + k, aBase + k + 16);
            bf16x16 bfr[8];
            #pragma unroll
            for (int nt = 0; nt < 8; ++nt) {
                // B fragment: lane's column = nt*16 + l, K in {8hh..8hh+7, 16+8hh..16+8hh+7}
                const bf16_t* bp = &wtLds[(nt * 16 + l) * SH + k + hh * 8];
                bfr[nt] = lds_frag(bp, bp + 16);
            }
            #pragma unroll
            for (int nt = 0; nt < 8; ++nt) {
                acc[nt] = __builtin_amdgcn_wmma_f32_16x16x32_bf16(
                    false, a, false, bfr[nt], (short)0, acc[nt], false, false);
            }
        }
    };

    // bias + LeakyReLU(0.2) on f32 accumulators, write bf16 activations.
    // C/D layout: VGPR r -> M = waveRow0 + hh*8 + r, N = nt*16 + l.
    auto store_h = [&]() {
        #pragma unroll
        for (int nt = 0; nt < 8; ++nt) {
            int n = nt * 16 + l;
            float bv = biasLds[n];
            #pragma unroll
            for (int r = 0; r < 8; ++r) {
                float v = acc[nt][r] + bv;
                v = (v >= 0.f) ? v : 0.2f * v;
                hLds[(waveRow0 + hh * 8 + r) * SH + n] = (bf16_t)v;
            }
        }
    };

    for (int i = 0; i < DDIM - 1; ++i) {
        if (i + 1 < DDIM - 1) {   // warm L2/WGP$ for the next iteration's big weights
            __builtin_prefetch(W2 + (size_t)(i + 1) * NH * NH, 0, 1);
            __builtin_prefetch(W3 + (size_t)(i + 1) * NH * NH, 0, 1);
        }

        // ---- layer 1: (B,32) x (32,128) ----
        __syncthreads();                                   // wt/st/h free to reuse
        load_wt(W1 + (size_t)i * DDIM * NH, DDIM, b1 + (size_t)i * NH);
        __syncthreads();
        #pragma unroll
        for (int nt = 0; nt < 8; ++nt) acc[nt] = vzero;
        mma(aX, DDIM);
        store_h();                                         // own rows only

        // ---- layer 2: (B,128) x (128,128) ----
        __syncthreads();
        load_wt(W2 + (size_t)i * NH * NH, NH, b2 + (size_t)i * NH);
        __syncthreads();
        #pragma unroll
        for (int nt = 0; nt < 8; ++nt) acc[nt] = vzero;
        mma(aH, NH);
        store_h();

        // ---- layer 3 (stage W4/b4 into LDS in the same phase) ----
        __syncthreads();
        load_wt(W3 + (size_t)i * NH * NH, NH, b3 + (size_t)i * NH);
        w4Lds[t] = W4[(size_t)i * NH * 2 + t];             // 256 floats
        if (t < 2) w4bLds[t] = b4[(size_t)i * 2 + t];
        __syncthreads();
        #pragma unroll
        for (int nt = 0; nt < 8; ++nt) acc[nt] = vzero;
        mma(aH, NH);
        store_h();

        // ---- layer 4: (B,128) x (128,2) — per-thread dots, all from LDS ----
        __syncthreads();                                   // h3 visible cross-wave
        {
            int r = t & 127, col = t >> 7;
            const bf16_t* hr = &hLds[r * SH];
            float acc4 = 0.f;
            #pragma unroll
            for (int k = 0; k < NH; k += 8) {
                bf16x8 hv = *(const bf16x8*)(hr + k);      // ds_load_b128
                #pragma unroll
                for (int j = 0; j < 8; ++j)
                    acc4 += (float)hv[j] * w4Lds[(k + j) * 2 + col];
            }
            stLds[r * 2 + col] = acc4 + w4bLds[col];
        }
        __syncthreads();

        if (t < BM) {
            float s  = stLds[t * 2 + 0];
            float tr = stLds[t * 2 + 1];
            int bg = row0 + t;
            float xv = x[(size_t)bg * DDIM + (i + 1)];     // f32 x for output precision
            z[(size_t)bg * DDIM + (i + 1)] = xv * __expf(s) + tr;
            ld_acc += s;
        }
    }

    // dim 0: learned constant (s0, t0) = p0
    if (t < BM) {
        int bg = row0 + t;
        float s0 = p0[0], t0 = p0[1];
        z[(size_t)bg * DDIM + 0] = x[(size_t)bg * DDIM + 0] * __expf(s0) + t0;
        logdet[bg] = ld_acc;                               // ld_acc started at p0[0]
    }
}

extern "C" void kernel_launch(void* const* d_in, const int* in_sizes, int n_in,
                              void* d_out, int out_size, void* d_ws, size_t ws_size,
                              hipStream_t stream) {
    const float* x  = (const float*)d_in[0];
    const float* p0 = (const float*)d_in[1];
    const float* W1 = (const float*)d_in[2];
    const float* b1 = (const float*)d_in[3];
    const float* W2 = (const float*)d_in[4];
    const float* b2 = (const float*)d_in[5];
    const float* W3 = (const float*)d_in[6];
    const float* b3 = (const float*)d_in[7];
    const float* W4 = (const float*)d_in[8];
    const float* b4 = (const float*)d_in[9];

    float* z      = (float*)d_out;                 // (B, D) floats
    float* logdet = z + (size_t)B_TOT * DDIM;      // (B,) floats

    dim3 grid(B_TOT / BM);
    dim3 block(NTHR);
    maf_fused_kernel<<<grid, block, 0, stream>>>(x, p0, W1, b1, W2, b2,
                                                 W3, b3, W4, b4, z, logdet);
}
